// CheckpointedIntegratedMTAPFMModel_24575802867932
// MI455X (gfx1250) — matrix operand order repaired and checked by reading, not compiled, not run on hardware
//
#include <hip/hip_runtime.h>
#include <hip/hip_bf16.h>

typedef __attribute__((ext_vector_type(16))) _Float16 v16h;
typedef __attribute__((ext_vector_type(8)))  _Float16 v8h;
typedef __attribute__((ext_vector_type(8)))  float    v8f;

#define HID    128
#define HSTR   136    /* padded h-staging row stride (f16): 272B -> conflict-free A loads */
#define GATES  512
#define BA_N   32768
#define T_ENC  20
#define T_DECN 12
#define FRAG_HALVES (32*4*32*16)   /* 65536 f16 per (512x128) weight matrix */

static_assert(FRAG_HALVES == 65536, "frag size");

#if defined(__has_builtin)
#if __has_builtin(__builtin_amdgcn_sched_barrier)
#define SCHED_FENCE() __builtin_amdgcn_sched_barrier(0)
#endif
#endif
#ifndef SCHED_FENCE
#define SCHED_FENCE()
#endif

// ---------------------------------------------------------------------------
// Fragment helpers
// ---------------------------------------------------------------------------
// A-matrix 16x32 f16 layout (ISA 7.12.2): lane l<16 holds row l, elems 0..7 =
// K..K+7, elems 8..15 = K+16..K+23 ; lane l+16 holds row l, K+8.. / K+24..
__device__ __forceinline__ v16h load_afrag(const _Float16* p) {
    v8h lo = *(const v8h*)p;
    v8h hp = *(const v8h*)(p + 16);
    v16h a;
#pragma unroll
    for (int i = 0; i < 8; ++i) { a[i] = lo[i]; a[i + 8] = hp[i]; }
    return a;
}

// B-fragment load with XOR bank swizzle.  Logical chunks (16B) of lane l are
// stored at chunk index (2l+h) ^ (l>>3): flips slot parity per lane octet so a
// 32-lane ds_load_b128 spreads over all 64 banks (2-way minimum conflict).
__device__ __forceinline__ v16h load_bfrag(const _Float16* fb, int group, int lane) {
    const v8h* p = (const v8h*)fb + group * 64;
    int c0 = (lane * 2) ^ (lane >> 3);
    v8h lo = p[c0];
    v8h hp = p[c0 ^ 1];
    v16h b;
#pragma unroll
    for (int i = 0; i < 8; ++i) { b[i] = lo[i]; b[i + 8] = hp[i]; }
    return b;
}

__device__ __forceinline__ float sigm(float x) {
    return 1.0f / (1.0f + __expf(-x));
}

// ---------------------------------------------------------------------------
// Weight pre-shuffle: W (512x128 f32, row = gate col) -> swizzled B-fragments.
// Logical: frag[group=ct*4+kc][lane][e] = W[col*128+k]
//   col = ct*16 + (lane&15) ; k = kc*32 + (e>>3)*16 + (lane>>4)*8 + (e&7)
// Physical 16B chunk index = (lane*2 + (e>>3)) ^ (lane>>3)  (see load_bfrag).
// ---------------------------------------------------------------------------
__global__ void make_frag_kernel(const float* __restrict__ W,
                                 _Float16* __restrict__ frag) {
    int idx  = blockIdx.x * 256 + threadIdx.x;      // 65536 total
    int e    = idx & 15;
    int lane = (idx >> 4) & 31;
    int kc   = (idx >> 9) & 3;
    int ct   = idx >> 11;
    int col  = ct * 16 + (lane & 15);
    int k    = kc * 32 + (e >> 3) * 16 + (lane >> 4) * 8 + (e & 7);
    int chunk = ((lane * 2 + (e >> 3)) ^ (lane >> 3));
    size_t phys = ((size_t)(ct * 4 + kc) * 64 + chunk) * 8 + (e & 7);
    frag[phys] = (_Float16)W[col * HID + k];
}

__global__ void bias_sum_kernel(const float* __restrict__ a,
                                const float* __restrict__ b,
                                float* __restrict__ out) {
    int i = blockIdx.x * 256 + threadIdx.x;
    if (i < GATES) out[i] = a[i] + b[i];
}

// g0base[c] = d_bih0[c] + d_bhh0[c] + sum_k b_embed[k] * d_wih0[c*128+k]
__global__ void g0base_kernel(const float* __restrict__ b_embed,
                              const float* __restrict__ d_wih0,
                              const float* __restrict__ d_bih0,
                              const float* __restrict__ d_bhh0,
                              float* __restrict__ out) {
    int c = blockIdx.x * 256 + threadIdx.x;
    if (c >= GATES) return;
    float acc = d_bih0[c] + d_bhh0[c];
    for (int k = 0; k < HID; ++k) acc += b_embed[k] * d_wih0[c * HID + k];
    out[c] = acc;
}

// x[t][row][c] = b_embed[c] + history[row][t][0]*W[c][0] + history[row][t][1]*W[c][1]
__global__ void embed_kernel(const float* __restrict__ history,
                             const float* __restrict__ w_embed,
                             const float* __restrict__ b_embed,
                             _Float16* __restrict__ x_out) {
    size_t idx = (size_t)blockIdx.x * 256 + threadIdx.x;
    size_t total = (size_t)T_ENC * BA_N * HID;
    if (idx >= total) return;
    int c      = (int)(idx & (HID - 1));
    size_t rt  = idx >> 7;
    int row    = (int)(rt % BA_N);
    int t      = (int)(rt / BA_N);
    float h0 = history[((size_t)row * T_ENC + t) * 2 + 0];
    float h1 = history[((size_t)row * T_ENC + t) * 2 + 1];
    x_out[idx] = (_Float16)(b_embed[c] + h0 * w_embed[c * 2] + h1 * w_embed[c * 2 + 1]);
}

// ---------------------------------------------------------------------------
// LSTM layer: block = 8 waves * 16 rows = 128 agents, loops all T steps.
// LDS: wih frags (128KB) + whh frags (128KB) + padded per-wave h staging.
// ---------------------------------------------------------------------------
__global__ __launch_bounds__(256)
void lstm_layer_kernel(const _Float16* x_seq,              // [T][BA][128] f16
                       const _Float16* __restrict__ wih_frag,
                       const _Float16* __restrict__ whh_frag,
                       const float*    __restrict__ bsum,   // [512]
                       _Float16* h_seq_out,                 // [T][BA][128] or null
                       float* __restrict__ hT_out,          // [BA][128] or null
                       float* __restrict__ cT_out,
                       int T) {
    extern __shared__ __attribute__((aligned(32))) _Float16 smem[];
    _Float16* s_wih = smem;
    _Float16* s_whh = smem + FRAG_HALVES;
    _Float16* s_h   = smem + 2 * FRAG_HALVES;   // 8 waves * 16*HSTR f16

    const int tid = threadIdx.x;
    {   // cooperative weight staging into LDS (one time)
        const uint4* s0 = (const uint4*)wih_frag;  uint4* d0 = (uint4*)s_wih;
        const uint4* s1 = (const uint4*)whh_frag;  uint4* d1 = (uint4*)s_whh;
        const int n16 = FRAG_HALVES * 2 / 16;      // 8192 uint4
        for (int i = tid; i < n16; i += 256) d0[i] = s0[i];
        for (int i = tid; i < n16; i += 256) d1[i] = s1[i];
    }
    const int lane = tid & 31;
    const int wave = tid >> 5;
    const int hi   = lane >> 4;
    const int ln   = lane & 15;
    const size_t waveRow = (size_t)blockIdx.x * 128 + wave * 16;
    _Float16* hbuf = s_h + wave * (16 * HSTR);
    // zero h staging
    for (int i = lane; i < 16 * HSTR / 2; i += 32) ((unsigned*)hbuf)[i] = 0u;

    float cacc[8][8];
#pragma unroll
    for (int jt = 0; jt < 8; ++jt)
#pragma unroll
        for (int r = 0; r < 8; ++r) cacc[jt][r] = 0.0f;
    __syncthreads();

    for (int t = 0; t < T; ++t) {
        // A fragments: K 0..127 from x_seq (global), K 128..255 from hbuf (LDS)
        v16h afrag[8];
        const _Float16* xrow = x_seq + ((size_t)t * BA_N + waveRow) * HID;
#pragma unroll
        for (int kc = 0; kc < 4; ++kc)
            afrag[kc] = load_afrag(xrow + (size_t)ln * HID + kc * 32 + hi * 8);
#pragma unroll
        for (int kc = 0; kc < 4; ++kc)
            afrag[4 + kc] = load_afrag(hbuf + ln * HSTR + kc * 32 + hi * 8);
        SCHED_FENCE();

#pragma unroll
        for (int jt = 0; jt < 8; ++jt) {
            float bi = bsum[       jt * 16 + ln];
            float bf = bsum[128  + jt * 16 + ln];
            float bg = bsum[256  + jt * 16 + ln];
            float bo = bsum[384  + jt * 16 + ln];
            v8f ai, af_, ag, ao;
#pragma unroll
            for (int i = 0; i < 8; ++i) { ai[i] = bi; af_[i] = bf; ag[i] = bg; ao[i] = bo; }
#pragma unroll
            for (int kc = 0; kc < 8; ++kc) {
                const _Float16* fb = (kc < 4) ? s_wih : s_whh;
                const int kcl = kc & 3;
                v16h a  = afrag[kc];
                v16h b0 = load_bfrag(fb, (jt     ) * 4 + kcl, lane);
                v16h b1 = load_bfrag(fb, (jt +  8) * 4 + kcl, lane);
                v16h b2 = load_bfrag(fb, (jt + 16) * 4 + kcl, lane);
                v16h b3 = load_bfrag(fb, (jt + 24) * 4 + kcl, lane);
                ai  = __builtin_amdgcn_wmma_f32_16x16x32_f16(false, a, false, b0, (short)0, ai,  false, false);
                af_ = __builtin_amdgcn_wmma_f32_16x16x32_f16(false, a, false, b1, (short)0, af_, false, false);
                ag  = __builtin_amdgcn_wmma_f32_16x16x32_f16(false, a, false, b2, (short)0, ag,  false, false);
                ao  = __builtin_amdgcn_wmma_f32_16x16x32_f16(false, a, false, b3, (short)0, ao,  false, false);
            }
#pragma unroll
            for (int r = 0; r < 8; ++r) {
                float iv = sigm(ai[r]);
                float fv = sigm(af_[r]);
                float gv = tanhf(ag[r]);
                float ov = sigm(ao[r]);
                float c2 = fv * cacc[jt][r] + iv * gv;
                cacc[jt][r] = c2;
                float h2 = ov * tanhf(c2);
                int rl = r + 8 * hi;
                hbuf[rl * HSTR + jt * 16 + ln] = (_Float16)h2;
                if (h_seq_out)
                    h_seq_out[((size_t)t * BA_N + waveRow + rl) * HID + jt * 16 + ln] = (_Float16)h2;
                if (hT_out && t == T - 1) {
                    hT_out[(waveRow + rl) * HID + jt * 16 + ln] = h2;
                    cT_out[(waveRow + rl) * HID + jt * 16 + ln] = c2;
                }
            }
            SCHED_FENCE();   // cap live ranges: don't interleave jt bodies
        }
        __syncthreads();
    }
}

// ---------------------------------------------------------------------------
// Fused 2-layer decoder, 12 steps. x_dec is constant -> layer0 input GEMM is
// the precomputed vector g0base.  LDS: whh0 + whh1 frags; wih1 streams from L2.
// Block = 4 waves * 16 rows = 64 agents.
// ---------------------------------------------------------------------------
__global__ __launch_bounds__(128)
void decoder_kernel(const float* __restrict__ hT, const float* __restrict__ cT,
                    const _Float16* __restrict__ whh0_frag,
                    const _Float16* __restrict__ wih1_frag,   // stays in global/L2
                    const _Float16* __restrict__ whh1_frag,
                    const float* __restrict__ g0base,         // [512]
                    const float* __restrict__ b1sum,          // [512]
                    const float* __restrict__ w_out,          // [2][128]
                    const float* __restrict__ b_out,          // [2]
                    const float* __restrict__ w_post,         // [2][2]
                    const float* __restrict__ b_post,         // [2]
                    float* __restrict__ preds) {              // [12][BA][2]
    extern __shared__ __attribute__((aligned(32))) _Float16 smem[];
    _Float16* s_whh0 = smem;
    _Float16* s_whh1 = smem + FRAG_HALVES;
    _Float16* s_h    = smem + 2 * FRAG_HALVES;   // 4 waves * 2 * 16*HSTR f16

    const int tid = threadIdx.x;
    {
        const uint4* s0 = (const uint4*)whh0_frag;  uint4* d0 = (uint4*)s_whh0;
        const uint4* s1 = (const uint4*)whh1_frag;  uint4* d1 = (uint4*)s_whh1;
        const int n16 = FRAG_HALVES * 2 / 16;
        for (int i = tid; i < n16; i += 128) d0[i] = s0[i];
        for (int i = tid; i < n16; i += 128) d1[i] = s1[i];
    }
    const int lane = tid & 31;
    const int wave = tid >> 5;
    const int hi   = lane >> 4;
    const int ln   = lane & 15;
    const size_t waveRow = (size_t)blockIdx.x * 64 + wave * 16;
    _Float16* h0buf = s_h + wave * (2 * 16 * HSTR);
    _Float16* h1buf = h0buf + 16 * HSTR;

    float c0acc[8][8], c1acc[8][8];
#pragma unroll
    for (int jt = 0; jt < 8; ++jt)
#pragma unroll
        for (int r = 0; r < 8; ++r) {
            int rl  = r + 8 * hi;
            int col = jt * 16 + ln;
            float hv = hT[(waveRow + rl) * HID + col];
            float cv = cT[(waveRow + rl) * HID + col];
            h0buf[rl * HSTR + col] = (_Float16)hv;
            h1buf[rl * HSTR + col] = (_Float16)hv;
            c0acc[jt][r] = cv;
            c1acc[jt][r] = cv;
        }
    __syncthreads();

    for (int t = 0; t < T_DECN; ++t) {
        // ---- layer 0: g0 = g0base + h0 @ whh0^T ----
        v16h a0[4];
#pragma unroll
        for (int kc = 0; kc < 4; ++kc)
            a0[kc] = load_afrag(h0buf + ln * HSTR + kc * 32 + hi * 8);
        SCHED_FENCE();
#pragma unroll
        for (int jt = 0; jt < 8; ++jt) {
            float bi = g0base[       jt * 16 + ln];
            float bf = g0base[128  + jt * 16 + ln];
            float bg = g0base[256  + jt * 16 + ln];
            float bo = g0base[384  + jt * 16 + ln];
            v8f ai, af_, ag, ao;
#pragma unroll
            for (int i = 0; i < 8; ++i) { ai[i] = bi; af_[i] = bf; ag[i] = bg; ao[i] = bo; }
#pragma unroll
            for (int kc = 0; kc < 4; ++kc) {
                v16h a  = a0[kc];
                v16h b0 = load_bfrag(s_whh0, (jt     ) * 4 + kc, lane);
                v16h b1 = load_bfrag(s_whh0, (jt +  8) * 4 + kc, lane);
                v16h b2 = load_bfrag(s_whh0, (jt + 16) * 4 + kc, lane);
                v16h b3 = load_bfrag(s_whh0, (jt + 24) * 4 + kc, lane);
                ai  = __builtin_amdgcn_wmma_f32_16x16x32_f16(false, a, false, b0, (short)0, ai,  false, false);
                af_ = __builtin_amdgcn_wmma_f32_16x16x32_f16(false, a, false, b1, (short)0, af_, false, false);
                ag  = __builtin_amdgcn_wmma_f32_16x16x32_f16(false, a, false, b2, (short)0, ag,  false, false);
                ao  = __builtin_amdgcn_wmma_f32_16x16x32_f16(false, a, false, b3, (short)0, ao,  false, false);
            }
#pragma unroll
            for (int r = 0; r < 8; ++r) {
                float iv = sigm(ai[r]);
                float fv = sigm(af_[r]);
                float gv = tanhf(ag[r]);
                float ov = sigm(ao[r]);
                float c2 = fv * c0acc[jt][r] + iv * gv;
                c0acc[jt][r] = c2;
                float h2 = ov * tanhf(c2);
                int rl = r + 8 * hi;
                h0buf[rl * HSTR + jt * 16 + ln] = (_Float16)h2;
            }
            SCHED_FENCE();
        }
        __syncthreads();

        // ---- layer 1: g1 = b1sum + h0n @ wih1^T + h1 @ whh1^T ----
        v16h a1[8];
#pragma unroll
        for (int kc = 0; kc < 4; ++kc)
            a1[kc] = load_afrag(h0buf + ln * HSTR + kc * 32 + hi * 8);
#pragma unroll
        for (int kc = 0; kc < 4; ++kc)
            a1[4 + kc] = load_afrag(h1buf + ln * HSTR + kc * 32 + hi * 8);
        SCHED_FENCE();
#pragma unroll
        for (int jt = 0; jt < 8; ++jt) {
            float bi = b1sum[       jt * 16 + ln];
            float bf = b1sum[128  + jt * 16 + ln];
            float bg = b1sum[256  + jt * 16 + ln];
            float bo = b1sum[384  + jt * 16 + ln];
            v8f ai, af_, ag, ao;
#pragma unroll
            for (int i = 0; i < 8; ++i) { ai[i] = bi; af_[i] = bf; ag[i] = bg; ao[i] = bo; }
#pragma unroll
            for (int kc = 0; kc < 8; ++kc) {
                const _Float16* fb = (kc < 4) ? wih1_frag : s_whh1;
                const int kcl = kc & 3;
                v16h a  = a1[kc];
                v16h b0 = load_bfrag(fb, (jt     ) * 4 + kcl, lane);
                v16h b1 = load_bfrag(fb, (jt +  8) * 4 + kcl, lane);
                v16h b2 = load_bfrag(fb, (jt + 16) * 4 + kcl, lane);
                v16h b3 = load_bfrag(fb, (jt + 24) * 4 + kcl, lane);
                ai  = __builtin_amdgcn_wmma_f32_16x16x32_f16(false, a, false, b0, (short)0, ai,  false, false);
                af_ = __builtin_amdgcn_wmma_f32_16x16x32_f16(false, a, false, b1, (short)0, af_, false, false);
                ag  = __builtin_amdgcn_wmma_f32_16x16x32_f16(false, a, false, b2, (short)0, ag,  false, false);
                ao  = __builtin_amdgcn_wmma_f32_16x16x32_f16(false, a, false, b3, (short)0, ao,  false, false);
            }
#pragma unroll
            for (int r = 0; r < 8; ++r) {
                float iv = sigm(ai[r]);
                float fv = sigm(af_[r]);
                float gv = tanhf(ag[r]);
                float ov = sigm(ao[r]);
                float c2 = fv * c1acc[jt][r] + iv * gv;
                c1acc[jt][r] = c2;
                float h2 = ov * tanhf(c2);
                int rl = r + 8 * hi;
                h1buf[rl * HSTR + jt * 16 + ln] = (_Float16)h2;
            }
            SCHED_FENCE();
        }
        __syncthreads();

        // ---- pred = (h1n @ w_out^T + b_out) @ w_post^T + b_post ----
        if (lane < 16) {
            int rl = lane;
            float p0 = b_out[0], p1 = b_out[1];
            for (int k = 0; k < HID; ++k) {
                float hv = (float)h1buf[rl * HSTR + k];
                p0 += hv * w_out[k];
                p1 += hv * w_out[HID + k];
            }
            float o0 = p0 * w_post[0] + p1 * w_post[1] + b_post[0];
            float o1 = p0 * w_post[2] + p1 * w_post[3] + b_post[1];
            size_t base = ((size_t)t * BA_N + waveRow + rl) * 2;
            preds[base + 0] = o0;
            preds[base + 1] = o1;
        }
        __syncthreads();
    }
}

// ---------------------------------------------------------------------------
// coeffs = hT @ w_coeff^T + b_coeff ; per-block deterministic partial sums
// ---------------------------------------------------------------------------
__global__ __launch_bounds__(256)
void coeffs_kernel(const float* __restrict__ hT,
                   const float* __restrict__ w_coeff,
                   const float* __restrict__ b_coeff,
                   float* __restrict__ coeffs,
                   float* __restrict__ partials, int nblocks) {
    __shared__ float ssum[256], ssq[256];
    int row = blockIdx.x * 256 + threadIdx.x;
    float lsum = 0.0f, lsq = 0.0f;
    if (row < BA_N) {
        for (int c = 0; c < 3; ++c) {
            float acc = b_coeff[c];
            for (int k = 0; k < HID; ++k) acc += hT[(size_t)row * HID + k] * w_coeff[c * HID + k];
            coeffs[row * 3 + c] = acc;
            lsum += acc;
            lsq  += acc * acc;
        }
    }
    ssum[threadIdx.x] = lsum;
    ssq[threadIdx.x]  = lsq;
    __syncthreads();
    for (int s = 128; s > 0; s >>= 1) {
        if (threadIdx.x < s) {
            ssum[threadIdx.x] += ssum[threadIdx.x + s];
            ssq[threadIdx.x]  += ssq[threadIdx.x + s];
        }
        __syncthreads();
    }
    if (threadIdx.x == 0) {
        partials[blockIdx.x]           = ssum[0];
        partials[nblocks + blockIdx.x] = ssq[0];
    }
}

__global__ void finalize_kernel(const float* __restrict__ partials, int nblocks,
                                float* __restrict__ out2, float invN) {
    if (threadIdx.x == 0 && blockIdx.x == 0) {
        float S = 0.0f, SS = 0.0f;
        for (int i = 0; i < nblocks; ++i) { S += partials[i]; SS += partials[nblocks + i]; }
        float mean = S * invN;
        out2[0] = mean;
        out2[1] = SS * invN - mean * mean;
    }
}

// ---------------------------------------------------------------------------
// Potential-field + speed-constraint scans, one thread per agent
// ---------------------------------------------------------------------------
#define REP_R   0.5f
#define TGT_SP  4.087f
#define MIN_SP  (TGT_SP * 0.85f)
#define MAX_SP  (TGT_SP * 1.15f)

__global__ __launch_bounds__(256)
void pfm_kernel(const float* __restrict__ history,
                const float* __restrict__ neighbors,
                const float* __restrict__ goal,
                const float* __restrict__ coeffs,
                const float* __restrict__ preds,
                float* __restrict__ outp) {
    int row = blockIdx.x * 256 + threadIdx.x;
    if (row >= BA_N) return;
    float k1 = coeffs[row * 3 + 0];
    float k2 = coeffs[row * 3 + 1];
    float kr = coeffs[row * 3 + 2];
    float gx = goal[0], gy = goal[1];
    float lx = history[((size_t)row * T_ENC + (T_ENC - 1)) * 2 + 0];
    float ly = history[((size_t)row * T_ENC + (T_ENC - 1)) * 2 + 1];
    const float* nb = neighbors + (size_t)row * 64 * 2;

    float ax[T_DECN], ay[T_DECN];
    float px = lx, py = ly;
    for (int t = 0; t < T_DECN; ++t) {
        float prx = preds[((size_t)t * BA_N + row) * 2 + 0];
        float pry = preds[((size_t)t * BA_N + row) * 2 + 1];
        float fx = k1 * (gx - px) + k2 * (prx - px);
        float fy = k1 * (gy - py) + k2 * (pry - py);
        float frx = 0.0f, fry = 0.0f;
        for (int n = 0; n < 64; ++n) {
            float dx = px - nb[n * 2 + 0];
            float dy = py - nb[n * 2 + 1];
            float dist = sqrtf(dx * dx + dy * dy) + 1e-6f;
            float m = (dist < REP_R) ? 1.0f : 0.0f;
            float s = kr * m / (dist * dist);
            frx += s * dx;
            fry += s * dy;
        }
        px += fx + frx;
        py += fy + fry;
        ax[t] = px;
        ay[t] = py;
    }
    px = lx; py = ly;
    for (int t = 0; t < T_DECN; ++t) {
        float dx = ax[t] - px, dy = ay[t] - py;
        float sp = sqrtf(dx * dx + dy * dy);
        float fin = (sp > 0.0f) ? fminf(fmaxf(sp, MIN_SP), MAX_SP) : sp;
        float inv = fin / (sp + 1e-8f);
        px += dx * inv;
        py += dy * inv;
        outp[((size_t)row * T_DECN + t) * 2 + 0] = px;
        outp[((size_t)row * T_DECN + t) * 2 + 1] = py;
    }
}

// ---------------------------------------------------------------------------
// Host side
// ---------------------------------------------------------------------------
extern "C" void kernel_launch(void* const* d_in, const int* in_sizes, int n_in,
                              void* d_out, int out_size, void* d_ws, size_t ws_size,
                              hipStream_t stream) {
    const float* history   = (const float*)d_in[0];
    const float* neighbors = (const float*)d_in[1];
    const float* goal      = (const float*)d_in[2];
    const float* w_embed   = (const float*)d_in[3];
    const float* b_embed   = (const float*)d_in[4];
    const float* e_wih0    = (const float*)d_in[5];
    const float* e_whh0    = (const float*)d_in[6];
    const float* e_bih0    = (const float*)d_in[7];
    const float* e_bhh0    = (const float*)d_in[8];
    const float* e_wih1    = (const float*)d_in[9];
    const float* e_whh1    = (const float*)d_in[10];
    const float* e_bih1    = (const float*)d_in[11];
    const float* e_bhh1    = (const float*)d_in[12];
    const float* d_wih0    = (const float*)d_in[13];
    const float* d_whh0    = (const float*)d_in[14];
    const float* d_bih0    = (const float*)d_in[15];
    const float* d_bhh0    = (const float*)d_in[16];
    const float* d_wih1    = (const float*)d_in[17];
    const float* d_whh1    = (const float*)d_in[18];
    const float* d_bih1    = (const float*)d_in[19];
    const float* d_bhh1    = (const float*)d_in[20];
    const float* w_out     = (const float*)d_in[21];
    const float* b_out     = (const float*)d_in[22];
    const float* w_post    = (const float*)d_in[23];
    const float* b_post    = (const float*)d_in[24];
    const float* w_coeff   = (const float*)d_in[25];
    const float* b_coeff   = (const float*)d_in[26];

    char* ws = (char*)d_ws;
    size_t off = 0;
    auto take = [&](size_t bytes) -> char* {
        char* p = ws + off;
        off = (off + bytes + 255) & ~(size_t)255;
        return p;
    };
    _Float16* xh        = (_Float16*)take((size_t)T_ENC * BA_N * HID * 2); // x_embed, reused in-place as hs0
    float*    hT        = (float*)take((size_t)BA_N * HID * 4);
    float*    cT        = (float*)take((size_t)BA_N * HID * 4);
    float*    preds     = (float*)take((size_t)T_DECN * BA_N * 2 * 4);
    float*    coeffs    = (float*)take((size_t)BA_N * 3 * 4);
    _Float16* fr_e0_wih = (_Float16*)take(FRAG_HALVES * 2);
    _Float16* fr_e0_whh = (_Float16*)take(FRAG_HALVES * 2);
    _Float16* fr_e1_wih = (_Float16*)take(FRAG_HALVES * 2);
    _Float16* fr_e1_whh = (_Float16*)take(FRAG_HALVES * 2);
    _Float16* fr_d0_whh = (_Float16*)take(FRAG_HALVES * 2);
    _Float16* fr_d1_wih = (_Float16*)take(FRAG_HALVES * 2);
    _Float16* fr_d1_whh = (_Float16*)take(FRAG_HALVES * 2);
    float*    bs_e0     = (float*)take(GATES * 4);
    float*    bs_e1     = (float*)take(GATES * 4);
    float*    g0base    = (float*)take(GATES * 4);
    float*    b1sum     = (float*)take(GATES * 4);
    float*    partials  = (float*)take(128 * 2 * 4);

    float* out_traj  = (float*)d_out;                 // [B,A,T_DEC,2]
    float* out_stats = out_traj + (size_t)BA_N * T_DECN * 2;

    // --- prep: embed, weight shuffles, bias precompute ---
    {
        size_t total = (size_t)T_ENC * BA_N * HID;
        embed_kernel<<<(unsigned)((total + 255) / 256), 256, 0, stream>>>(history, w_embed, b_embed, xh);
    }
    make_frag_kernel<<<FRAG_HALVES / 256, 256, 0, stream>>>(e_wih0, fr_e0_wih);
    make_frag_kernel<<<FRAG_HALVES / 256, 256, 0, stream>>>(e_whh0, fr_e0_whh);
    make_frag_kernel<<<FRAG_HALVES / 256, 256, 0, stream>>>(e_wih1, fr_e1_wih);
    make_frag_kernel<<<FRAG_HALVES / 256, 256, 0, stream>>>(e_whh1, fr_e1_whh);
    make_frag_kernel<<<FRAG_HALVES / 256, 256, 0, stream>>>(d_whh0, fr_d0_whh);
    make_frag_kernel<<<FRAG_HALVES / 256, 256, 0, stream>>>(d_wih1, fr_d1_wih);
    make_frag_kernel<<<FRAG_HALVES / 256, 256, 0, stream>>>(d_whh1, fr_d1_whh);
    bias_sum_kernel<<<2, 256, 0, stream>>>(e_bih0, e_bhh0, bs_e0);
    bias_sum_kernel<<<2, 256, 0, stream>>>(e_bih1, e_bhh1, bs_e1);
    bias_sum_kernel<<<2, 256, 0, stream>>>(d_bih1, d_bhh1, b1sum);
    g0base_kernel<<<2, 256, 0, stream>>>(b_embed, d_wih0, d_bih0, d_bhh0, g0base);

    // --- encoder: 2 LSTM layers over 20 steps (layer0 writes hs0 in place) ---
    const size_t lstm_lds = (size_t)2 * FRAG_HALVES * 2 + 8 * 16 * HSTR * 2;  // ~290KB
    lstm_layer_kernel<<<BA_N / 128, 256, lstm_lds, stream>>>(
        xh, fr_e0_wih, fr_e0_whh, bs_e0, xh, nullptr, nullptr, T_ENC);
    lstm_layer_kernel<<<BA_N / 128, 256, lstm_lds, stream>>>(
        xh, fr_e1_wih, fr_e1_whh, bs_e1, nullptr, hT, cT, T_ENC);

    // --- fused 2-layer decoder (12 steps) ---
    const size_t dec_lds = (size_t)2 * FRAG_HALVES * 2 + 4 * 2 * 16 * HSTR * 2; // ~290KB
    decoder_kernel<<<BA_N / 64, 128, dec_lds, stream>>>(
        hT, cT, fr_d0_whh, fr_d1_wih, fr_d1_whh, g0base, b1sum,
        w_out, b_out, w_post, b_post, preds);

    // --- coeffs + deterministic mean/var ---
    coeffs_kernel<<<BA_N / 256, 256, 0, stream>>>(hT, w_coeff, b_coeff, coeffs, partials, BA_N / 256);
    finalize_kernel<<<1, 64, 0, stream>>>(partials, BA_N / 256, out_stats, 1.0f / (float)(BA_N * 3));

    // --- potential-field + speed constraint scans ---
    pfm_kernel<<<BA_N / 256, 256, 0, stream>>>(history, neighbors, goal, coeffs, preds, out_traj);
}
